// SwinBlock_78623671320896
// MI455X (gfx1250) — compile-verified, matrix operand order checked by reading, not run on hardware
//
#include <hip/hip_runtime.h>
#include <hip/hip_bf16.h>

typedef __attribute__((ext_vector_type(16))) _Float16 v16h;
typedef __attribute__((ext_vector_type(8)))  float    v8f;

#define WS_   7
#define SHIFT_ 3
#define DIM_  192
#define HEADS_ 6
#define HD_   32
#define NTOK_ 49      // tokens per window
#define NPAD_ 64      // padded rows
#define NWIN_ 2048    // B * (H/WS)*(W/WS) = 32*64
#define HW_   3136
#define B_    32
#define HIDDEN_ 768
#define TOKENS_ 100352  // 32*3136

// ---------------------------------------------------------------------------
// CDNA5 async copy: global -> LDS, 16 bytes per lane, tracked by ASYNCcnt.
// GVS addressing: mem = SADDR(64b, uniform) + VADDR(32b, per-lane) ; LDS dest
// address is a per-lane byte offset in VDST.
// ---------------------------------------------------------------------------
__device__ inline void async_b128(unsigned lds_off, unsigned gv_off, const void* sbase) {
  asm volatile("global_load_async_to_lds_b128 %0, %1, %2"
               :: "v"(lds_off), "v"(gv_off), "s"(sbase)
               : "memory");
}
__device__ inline void async_wait0() {
  asm volatile("s_wait_asynccnt 0" ::: "memory");
}
__device__ inline unsigned lds_addr_of(const void* p) {
  return (unsigned)(unsigned long long)p;   // low 32 bits of generic LDS ptr = LDS offset
}

// ---------------------------------------------------------------------------
// Fragment loader: 16-bit A/B operand for v_wmma_f32_16x16x32_f16.
// ISA layout (16-bit A 16x32): lanes 0-15 hold M=0..15 with K=0..7 in v0..3,
// K=16..23 in v4..7; lanes 16-31 hold M=0..15 with K=8..15 / K=24..31.
// B operand uses the mirrored layout with N in place of M, so loading the
// pre-transposed Bt (row-major [N][K]) with the same loader is correct.
// Two 16-byte loads per fragment (contiguous K runs of 8 halves); works for
// both global and LDS (generic) pointers.
// ---------------------------------------------------------------------------
__device__ inline v16h load_frag(const _Float16* __restrict__ base, int ld) {
  int lane = threadIdx.x & 31;
  int m  = lane & 15;
  int kb = (lane >> 4) << 3;   // 0 or 8
  const _Float16* p = base + (size_t)m * ld + kb;
  v16h f;
  uint4* fp = reinterpret_cast<uint4*>(&f);
  fp[0] = *reinterpret_cast<const uint4*>(p);        // K = kb .. kb+7
  fp[1] = *reinterpret_cast<const uint4*>(p + 16);   // K = kb+16 .. kb+23
  return f;
}

// Accumulate one 16x16 tile: A row-major (lda), Bt row-major [N][K] (ldb), K%32==0.
__device__ inline v8f wmma_kloop(const _Float16* __restrict__ A, int lda,
                                 const _Float16* __restrict__ Bt, int ldb, int K,
                                 v8f acc) {
  for (int k0 = 0; k0 < K; k0 += 32) {
    v16h a = load_frag(A + k0, lda);
    v16h b = load_frag(Bt + k0, ldb);
    acc = __builtin_amdgcn_wmma_f32_16x16x32_f16(false, a, false, b,
                                                 (short)0, acc, false, false);
  }
  return acc;
}

// ---------------------------------------------------------------------------
// K0: fp32 weight (K x N, row-major) -> f16 transposed (N x K, row-major)
// ---------------------------------------------------------------------------
__global__ void k_wt(const float* __restrict__ w, _Float16* __restrict__ wt,
                     int K, int N) {
  int id = blockIdx.x * 256 + threadIdx.x;
  if (id >= K * N) return;
  int k = id / N, n = id - k * N;
  wt[(size_t)n * K + k] = (_Float16)w[id];
}

// ---------------------------------------------------------------------------
// K1: LayerNorm1 + roll(-3,-3) + window partition -> winX f16 [win][64][192]
// ---------------------------------------------------------------------------
__global__ void k_ln1_window(const float* __restrict__ x, const float* __restrict__ g1,
                             const float* __restrict__ b1, _Float16* __restrict__ winX) {
  int win = blockIdx.x;
  int wid = threadIdx.x >> 5, lane = threadIdx.x & 31;
  int b = win >> 6, slot = win & 63, wy = slot >> 3, wx = slot & 7;
  for (int n = wid; n < NPAD_; n += 8) {
    _Float16* dst = winX + ((size_t)win * NPAD_ + n) * DIM_;
    if (n >= NTOK_) {
      for (int c = lane; c < DIM_; c += 32) dst[c] = (_Float16)0.f;
      continue;
    }
    int ny = n / WS_, nx = n - ny * WS_;
    int hh = (wy * WS_ + ny + SHIFT_) % 56;
    int ww = (wx * WS_ + nx + SHIFT_) % 56;
    const float* src = x + ((size_t)b * HW_ + hh * 56 + ww) * DIM_;
    float s = 0.f, ss = 0.f, v[6];
#pragma unroll
    for (int i = 0; i < 6; ++i) { float t = src[lane + 32 * i]; v[i] = t; s += t; ss += t * t; }
#pragma unroll
    for (int m = 16; m; m >>= 1) { s += __shfl_xor(s, m, 32); ss += __shfl_xor(ss, m, 32); }
    float mu  = s * (1.f / DIM_);
    float var = ss * (1.f / DIM_) - mu * mu;
    float inv = rsqrtf(var + 1e-5f);
#pragma unroll
    for (int i = 0; i < 6; ++i) {
      int c = lane + 32 * i;
      dst[c] = (_Float16)((v[i] - mu) * inv * g1[c] + b1[c]);
    }
  }
}

// ---------------------------------------------------------------------------
// K2: per-window QKV GEMM (64x192 @ 192x576) + bias.
// A window (24.5KB, contiguous) async-staged to LDS once per block.
// ---------------------------------------------------------------------------
__global__ void k_qkv(const _Float16* __restrict__ winX, const _Float16* __restrict__ qkvT,
                      const float* __restrict__ qkv_b, _Float16* __restrict__ q,
                      _Float16* __restrict__ kmat, _Float16* __restrict__ vT) {
  __shared__ __align__(16) _Float16 sA[NPAD_ * DIM_];   // 24576 B
  int win = blockIdx.x;
  int wid = threadIdx.x >> 5, lane = threadIdx.x & 31;
  int cn = lane & 15, rb = (lane >> 4) * 8;

  { // async stage A
    const void* gbase = (const void*)(winX + (size_t)win * NPAD_ * DIM_);
    unsigned lbase = lds_addr_of(sA);
#pragma unroll
    for (int i = 0; i < 6; ++i) {
      unsigned c = (threadIdx.x + 256 * i) * 16;   // 1536 chunks of 16B
      async_b128(lbase + c, c, gbase);
    }
    async_wait0();
    __syncthreads();
  }

  for (int t = wid; t < 4 * 36; t += 8) {
    int ti = t / 36, tj = t - ti * 36;
    v8f acc = {};
    acc = wmma_kloop(sA + (size_t)(ti * 16) * DIM_, DIM_,
                     qkvT + (size_t)(tj * 16) * DIM_, DIM_, DIM_, acc);
    int j = tj * 16 + cn;               // 0..575
    int part = j / DIM_, jj = j - part * DIM_;
    int head = jj >> 5, d = jj & 31;
    float bj = qkv_b[j];
    size_t wh = (size_t)win * HEADS_ + head;
#pragma unroll
    for (int r = 0; r < 8; ++r) {
      int m = ti * 16 + rb + r;
      float val = acc[r] + bj;
      if (part == 0)      q[(wh * NPAD_ + m) * HD_ + d] = (_Float16)(val * 0.17677669529663687f);
      else if (part == 1) kmat[(wh * NPAD_ + m) * HD_ + d] = (_Float16)val;
      else                vT[(wh * HD_ + d) * NPAD_ + m] = (_Float16)(m < NTOK_ ? val : 0.f);
    }
  }
}

// ---------------------------------------------------------------------------
// K3: attention per (window, head): S = qk^T + relpos + mask; softmax; O = P@V.
// q/k/v blocks (4KB each) async-staged to LDS.
// ---------------------------------------------------------------------------
__global__ void k_attn(const _Float16* __restrict__ q, const _Float16* __restrict__ kmat,
                       const _Float16* __restrict__ vT, const float* __restrict__ rel_table,
                       const float* __restrict__ amask, _Float16* __restrict__ attnO) {
  __shared__ __align__(16) _Float16 sQ[NPAD_ * HD_];    // 4096 B
  __shared__ __align__(16) _Float16 sK[NPAD_ * HD_];    // 4096 B
  __shared__ __align__(16) _Float16 sV[HD_ * NPAD_];    // 4096 B
  __shared__ float    S[NPAD_][NPAD_];
  __shared__ _Float16 P[NPAD_][NPAD_];
  int win = blockIdx.x, head = blockIdx.y;
  size_t wh = (size_t)win * HEADS_ + head;
  int wid = threadIdx.x >> 5, lane = threadIdx.x & 31;
  int cn = lane & 15, rb = (lane >> 4) * 8;

  { // async stage q, k, v (each contiguous 4096 B -> 256 chunks of 16B)
    unsigned c = threadIdx.x * 16;
    async_b128(lds_addr_of(sQ) + c, c, (const void*)(q    + wh * NPAD_ * HD_));
    async_b128(lds_addr_of(sK) + c, c, (const void*)(kmat + wh * NPAD_ * HD_));
    async_b128(lds_addr_of(sV) + c, c, (const void*)(vT   + wh * HD_ * NPAD_));
    async_wait0();
    __syncthreads();
  }

  // ---- S tiles: 4x4, one K=32 WMMA each ----
  for (int t = wid; t < 16; t += 8) {
    int ti = t >> 2, tj = t & 3;
    v16h a = load_frag(sQ + (size_t)(ti * 16) * HD_, HD_);
    v16h b = load_frag(sK + (size_t)(tj * 16) * HD_, HD_);
    v8f acc = {};
    acc = __builtin_amdgcn_wmma_f32_16x16x32_f16(false, a, false, b, (short)0, acc, false, false);
    int n = tj * 16 + cn;
#pragma unroll
    for (int r = 0; r < 8; ++r) {
      int m = ti * 16 + rb + r;
      float val;
      if (m < NTOK_ && n < NTOK_) {
        int yi = m / WS_, xi = m - yi * WS_;
        int yj = n / WS_, xj = n - yj * WS_;
        int ridx = (yi - yj + WS_ - 1) * (2 * WS_ - 1) + (xi - xj + WS_ - 1);
        val = acc[r] + rel_table[ridx * HEADS_ + head]
                     + amask[((size_t)(win & 63) * NTOK_ + m) * NTOK_ + n];
      } else {
        val = -1e30f;
      }
      S[m][n] = val;
    }
  }
  __syncthreads();

  // ---- row softmax (one wave per row, 2 cols per lane) ----
  for (int i = wid; i < NPAD_; i += 8) {
    float v0 = S[i][lane], v1 = S[i][lane + 32];
    float mx = fmaxf(v0, v1);
#pragma unroll
    for (int m = 16; m; m >>= 1) mx = fmaxf(mx, __shfl_xor(mx, m, 32));
    float e0 = __expf(v0 - mx), e1 = __expf(v1 - mx);
    float s = e0 + e1;
#pragma unroll
    for (int m = 16; m; m >>= 1) s += __shfl_xor(s, m, 32);
    float inv = 1.f / s;
    P[i][lane]      = (_Float16)(e0 * inv);
    P[i][lane + 32] = (_Float16)(e1 * inv);
  }
  __syncthreads();

  // ---- O = P(64x64) @ V(64x32): tiles 4x2, K=64 ----
  for (int t = wid; t < 8; t += 8) {
    int ti = t >> 1, tj = t & 1;
    v8f acc = {};
#pragma unroll
    for (int k0 = 0; k0 < NPAD_; k0 += 32) {
      v16h a = load_frag(&P[ti * 16][k0], NPAD_);
      v16h b = load_frag(sV + (size_t)(tj * 16) * NPAD_ + k0, NPAD_);
      acc = __builtin_amdgcn_wmma_f32_16x16x32_f16(false, a, false, b, (short)0, acc, false, false);
    }
    int d = tj * 16 + cn;
#pragma unroll
    for (int r = 0; r < 8; ++r) {
      int m = ti * 16 + rb + r;
      attnO[((size_t)win * NPAD_ + m) * DIM_ + head * HD_ + d] = (_Float16)acc[r];
    }
  }
}

// ---------------------------------------------------------------------------
// K4: proj GEMM + bias + reverse roll/window scatter + residual -> x2 (f32)
// A window async-staged to LDS once per block.
// ---------------------------------------------------------------------------
__global__ void k_proj(const _Float16* __restrict__ attnO, const _Float16* __restrict__ projT,
                       const float* __restrict__ proj_b, const float* __restrict__ x,
                       float* __restrict__ x2) {
  __shared__ __align__(16) _Float16 sA[NPAD_ * DIM_];   // 24576 B
  int win = blockIdx.x;
  int wid = threadIdx.x >> 5, lane = threadIdx.x & 31;
  int cn = lane & 15, rb = (lane >> 4) * 8;
  int b = win >> 6, slot = win & 63, wy = slot >> 3, wx = slot & 7;

  { // async stage A
    const void* gbase = (const void*)(attnO + (size_t)win * NPAD_ * DIM_);
    unsigned lbase = lds_addr_of(sA);
#pragma unroll
    for (int i = 0; i < 6; ++i) {
      unsigned c = (threadIdx.x + 256 * i) * 16;
      async_b128(lbase + c, c, gbase);
    }
    async_wait0();
    __syncthreads();
  }

  for (int t = wid; t < 4 * 12; t += 8) {
    int ti = t / 12, tj = t - ti * 12;
    v8f acc = {};
    acc = wmma_kloop(sA + (size_t)(ti * 16) * DIM_, DIM_,
                     projT + (size_t)(tj * 16) * DIM_, DIM_, DIM_, acc);
    int c = tj * 16 + cn;
    float pb = proj_b[c];
#pragma unroll
    for (int r = 0; r < 8; ++r) {
      int n = ti * 16 + rb + r;
      if (n >= NTOK_) continue;
      int ny = n / WS_, nx = n - ny * WS_;
      int hh = (wy * WS_ + ny + SHIFT_) % 56;
      int ww = (wx * WS_ + nx + SHIFT_) % 56;
      size_t idx = ((size_t)b * HW_ + hh * 56 + ww) * DIM_ + c;
      x2[idx] = x[idx] + acc[r] + pb;
    }
  }
}

// ---------------------------------------------------------------------------
// K5: LayerNorm2 -> h2 f16 (one wave per token)
// ---------------------------------------------------------------------------
__global__ void k_ln2(const float* __restrict__ x2, const float* __restrict__ g2,
                      const float* __restrict__ b2, _Float16* __restrict__ h2) {
  int tok = blockIdx.x * 8 + (threadIdx.x >> 5);
  int lane = threadIdx.x & 31;
  const float* src = x2 + (size_t)tok * DIM_;
  float s = 0.f, ss = 0.f, v[6];
#pragma unroll
  for (int i = 0; i < 6; ++i) { float t = src[lane + 32 * i]; v[i] = t; s += t; ss += t * t; }
#pragma unroll
  for (int m = 16; m; m >>= 1) { s += __shfl_xor(s, m, 32); ss += __shfl_xor(ss, m, 32); }
  float mu  = s * (1.f / DIM_);
  float var = ss * (1.f / DIM_) - mu * mu;
  float inv = rsqrtf(var + 1e-5f);
  _Float16* dst = h2 + (size_t)tok * DIM_;
#pragma unroll
  for (int i = 0; i < 6; ++i) {
    int c = lane + 32 * i;
    dst[c] = (_Float16)((v[i] - mu) * inv * g2[c] + b2[c]);
  }
}

// ---------------------------------------------------------------------------
// K6: FC1 GEMM (100352x192 @ 192x768) + bias + exact GELU -> f16
// A block (64x192, contiguous 24.5KB) async-staged to LDS.
// ---------------------------------------------------------------------------
__global__ void k_fc1(const _Float16* __restrict__ h2, const _Float16* __restrict__ fc1T,
                      const float* __restrict__ fc1_b, _Float16* __restrict__ fc1o) {
  __shared__ __align__(16) _Float16 sA[64 * DIM_];      // 24576 B
  int mb = blockIdx.x * 64, nb = blockIdx.y * 64;
  int wid = threadIdx.x >> 5, lane = threadIdx.x & 31;
  int cn = lane & 15, rb = (lane >> 4) * 8;

  { // async stage A (contiguous: row stride == DIM_)
    const void* gbase = (const void*)(h2 + (size_t)mb * DIM_);
    unsigned lbase = lds_addr_of(sA);
#pragma unroll
    for (int i = 0; i < 6; ++i) {
      unsigned c = (threadIdx.x + 256 * i) * 16;
      async_b128(lbase + c, c, gbase);
    }
    async_wait0();
    __syncthreads();
  }

  for (int t = wid * 2; t < wid * 2 + 2; ++t) {
    int ti = t >> 2, tj = t & 3;
    v8f acc = {};
    acc = wmma_kloop(sA + (size_t)(ti * 16) * DIM_, DIM_,
                     fc1T + (size_t)(nb + tj * 16) * DIM_, DIM_, DIM_, acc);
    int col = nb + tj * 16 + cn;
    float bb = fc1_b[col];
#pragma unroll
    for (int r = 0; r < 8; ++r) {
      int row = mb + ti * 16 + rb + r;
      float val = acc[r] + bb;
      val = val * 0.5f * (1.f + erff(val * 0.7071067811865475f));
      fc1o[(size_t)row * HIDDEN_ + col] = (_Float16)val;
    }
  }
}

// ---------------------------------------------------------------------------
// K7: FC2 GEMM (100352x768 @ 768x192) + bias + residual -> d_out f32
// K staged in three 64x256 LDS chunks (32KB), accumulators carried across.
// ---------------------------------------------------------------------------
__global__ void k_fc2(const _Float16* __restrict__ fc1o, const _Float16* __restrict__ fc2T,
                      const float* __restrict__ fc2_b, const float* __restrict__ x2,
                      float* __restrict__ out) {
  const int KC = 256;                                   // K chunk
  __shared__ __align__(16) _Float16 sA[64 * KC];        // 32768 B
  int mb = blockIdx.x * 64, nb = blockIdx.y * 64;
  int wid = threadIdx.x >> 5, lane = threadIdx.x & 31;
  int cn = lane & 15, rb = (lane >> 4) * 8;
  int t0 = wid * 2, t1 = wid * 2 + 1;
  int ti0 = t0 >> 2, tj0 = t0 & 3, ti1 = t1 >> 2, tj1 = t1 & 3;

  v8f acc0 = {}, acc1 = {};
  for (int kc = 0; kc < HIDDEN_; kc += KC) {
    { // async stage 64 x KC sub-block (row stride HIDDEN_ halves)
      const void* gbase = (const void*)(fc1o + (size_t)mb * HIDDEN_ + kc);
      unsigned lbase = lds_addr_of(sA);
#pragma unroll
      for (int i = 0; i < 8; ++i) {
        int c = threadIdx.x + 256 * i;      // 2048 chunks of 16B
        int row = c >> 5, colc = c & 31;    // 32 chunks per row
        async_b128(lbase + (unsigned)c * 16,
                   (unsigned)(row * (HIDDEN_ * 2) + colc * 16), gbase);
      }
      async_wait0();
      __syncthreads();
    }
    acc0 = wmma_kloop(sA + (size_t)(ti0 * 16) * KC, KC,
                      fc2T + (size_t)(nb + tj0 * 16) * HIDDEN_ + kc, HIDDEN_, KC, acc0);
    acc1 = wmma_kloop(sA + (size_t)(ti1 * 16) * KC, KC,
                      fc2T + (size_t)(nb + tj1 * 16) * HIDDEN_ + kc, HIDDEN_, KC, acc1);
    __syncthreads();    // before next chunk overwrites sA
  }

  float bb0 = fc2_b[nb + tj0 * 16 + cn];
  float bb1 = fc2_b[nb + tj1 * 16 + cn];
#pragma unroll
  for (int r = 0; r < 8; ++r) {
    int row0 = mb + ti0 * 16 + rb + r;
    size_t idx0 = (size_t)row0 * DIM_ + nb + tj0 * 16 + cn;
    out[idx0] = x2[idx0] + acc0[r] + bb0;
    int row1 = mb + ti1 * 16 + rb + r;
    size_t idx1 = (size_t)row1 * DIM_ + nb + tj1 * 16 + cn;
    out[idx1] = x2[idx1] + acc1[r] + bb1;
  }
}

// ---------------------------------------------------------------------------
extern "C" void kernel_launch(void* const* d_in, const int* in_sizes, int n_in,
                              void* d_out, int out_size, void* d_ws, size_t ws_size,
                              hipStream_t stream) {
  const float* x       = (const float*)d_in[0];
  const float* g1      = (const float*)d_in[1];
  const float* b1      = (const float*)d_in[2];
  const float* qkv_w   = (const float*)d_in[3];
  const float* qkv_b   = (const float*)d_in[4];
  const float* proj_w  = (const float*)d_in[5];
  const float* proj_b  = (const float*)d_in[6];
  const float* rel_tab = (const float*)d_in[7];
  const float* g2      = (const float*)d_in[8];
  const float* b2      = (const float*)d_in[9];
  const float* fc1_w   = (const float*)d_in[10];
  const float* fc1_b   = (const float*)d_in[11];
  const float* fc2_w   = (const float*)d_in[12];
  const float* fc2_b   = (const float*)d_in[13];
  const float* amask   = (const float*)d_in[14];

  // ---- workspace layout (256B aligned) ----
  char* ws = (char*)d_ws;
  size_t off = 0;
  auto take = [&](size_t bytes) -> char* {
    char* p = ws + off;
    off = (off + bytes + 255) & ~(size_t)255;
    return p;
  };
  _Float16* qkvT = (_Float16*)take((size_t)576 * 192 * 2);
  _Float16* projT = (_Float16*)take((size_t)192 * 192 * 2);
  _Float16* fc1T = (_Float16*)take((size_t)768 * 192 * 2);
  _Float16* fc2T = (_Float16*)take((size_t)192 * 768 * 2);
  float*    x2   = (float*)   take((size_t)TOKENS_ * DIM_ * 4);
  _Float16* h2   = (_Float16*)take((size_t)TOKENS_ * DIM_ * 2);
  // pool (phase 1): winX | q | k | vT | attnO ; fc1o (phase 2) aliases winX..vT
  const size_t SEG = (size_t)NWIN_ * NPAD_ * DIM_ * 2;   // 50,331,648 B
  char* pool = take(5 * SEG);
  _Float16* winX  = (_Float16*)(pool);
  _Float16* qbuf  = (_Float16*)(pool + 1 * SEG);
  _Float16* kbuf  = (_Float16*)(pool + 2 * SEG);
  _Float16* vT    = (_Float16*)(pool + 3 * SEG);
  _Float16* attnO = (_Float16*)(pool + 4 * SEG);
  _Float16* fc1o  = (_Float16*)(pool);                   // dead-region alias

  // K0: weight convert + transpose (fp32 KxN -> f16 NxK)
  k_wt<<<(192 * 576 + 255) / 256, 256, 0, stream>>>(qkv_w, qkvT, 192, 576);
  k_wt<<<(192 * 192 + 255) / 256, 256, 0, stream>>>(proj_w, projT, 192, 192);
  k_wt<<<(192 * 768 + 255) / 256, 256, 0, stream>>>(fc1_w, fc1T, 192, 768);
  k_wt<<<(768 * 192 + 255) / 256, 256, 0, stream>>>(fc2_w, fc2T, 768, 192);

  // K1: LN1 + shift + window partition
  k_ln1_window<<<NWIN_, 256, 0, stream>>>(x, g1, b1, winX);
  // K2: QKV GEMM
  k_qkv<<<NWIN_, 256, 0, stream>>>(winX, qkvT, qkv_b, qbuf, kbuf, vT);
  // K3: attention
  k_attn<<<dim3(NWIN_, HEADS_), 256, 0, stream>>>(qbuf, kbuf, vT, rel_tab, amask, attnO);
  // K4: proj + reverse shift + residual
  k_proj<<<NWIN_, 256, 0, stream>>>(attnO, projT, proj_b, x, x2);
  // K5: LN2
  k_ln2<<<TOKENS_ / 8, 256, 0, stream>>>(x2, g2, b2, h2);
  // K6: FC1 + GELU
  k_fc1<<<dim3(TOKENS_ / 64, HIDDEN_ / 64), 256, 0, stream>>>(h2, fc1T, fc1_b, fc1o);
  // K7: FC2 + residual -> out
  k_fc2<<<dim3(TOKENS_ / 64, DIM_ / 64), 256, 0, stream>>>(fc1o, fc2T, fc2_b, x2, (float*)d_out);
}
